// RenderNet_40389872452231
// MI455X (gfx1250) — compile-verified, matrix operand order checked by reading, not run on hardware
//
#include <hip/hip_runtime.h>
#include <hip/hip_bf16.h>

typedef __attribute__((ext_vector_type(16))) _Float16 v16h;
typedef __attribute__((ext_vector_type(8)))  float    v8f;

#define N_IM     2048
#define NN       (N_IM * N_IM)
#define PAD      3
#define TPW      2056              // padded template width (2048 + 6, rounded to 8)
#define TPH      2054              // padded template height (2048 + 6)
#define TP_WORDS ((TPH * TPW) / 2) // u32 words to clear (f16 template)
#define PATCH_H  38                // 32 output rows + 6 halo
#define PATCH_W  70                // 64 output cols + 6 halo
#define PATCH_DW (PATCH_W / 2)     // dwords per patch row (35)
#define LDSW     72                // padded LDS row stride (halfs)
#define KTOT     154               // 7 dy * 22 cols
#define NCHUNK   5                 // ceil(154/32)

union V16 { v16h v; _Float16 h[16]; };

// ---------------------------------------------------------------- init ----
__global__ void rn_init_kernel(unsigned* tp32, unsigned* mm) {
    long i = (long)blockIdx.x * blockDim.x + threadIdx.x;
    if (i < TP_WORDS) tp32[i] = 0u;
    if (blockIdx.x == 0 && threadIdx.x == 0) {
        mm[0] = 0x7F800000u;  // +inf  (min accumulator, nonneg-float bit order)
        mm[1] = 0x00000000u;  // 0     (max accumulator)
    }
}

// ---------------------------------------------------------- Bresenham ----
__global__ void rn_line_kernel(const float* px0, const float* py0,
                               const float* px1, const float* py1,
                               _Float16* tp) {
    float X0 = px0[0], Y0 = py0[0], X1 = px1[0], Y1 = py1[0];
    float adx = fabsf(X1 - X0), ady = fabsf(Y1 - Y0);
    float sx = (X1 > X0) ? 1.f : -1.f;
    float sy = (Y1 > Y0) ? 1.f : -1.f;
    bool steep = ady > adx;
    float x  = steep ? Y0 : X0;
    float y  = steep ? X0 : Y0;
    float DX = steep ? ady : adx;
    float DY = steep ? adx : ady;
    float SX = steep ? sy : sx;
    float SY = steep ? sx : sy;
    int n_iter = (int)floorf(DX);
    float d = 2.f * DY - DX;
    for (int i = 0; i < N_IM; ++i) {
        if (i < n_iter) {
            // reference: idx = n*row + col computed in f32 then truncated
            float fidx = steep ? (2048.f * x + y) : (2048.f * y + x);
            int idx = (int)fidx;
            idx = min(max(idx, 0), NN - 1);
            int r = idx >> 11, c = idx & (N_IM - 1);
            tp[(size_t)(r + PAD) * TPW + (c + PAD)] = (_Float16)1.f;
        }
        float mv = (d >= 0.f) ? 1.f : 0.f;
        y += SY * mv;
        d -= 2.f * DX * mv;
        x += SX;
        d += 2.f * DY;
    }
    int ei = (int)(2048.f * Y1 + X1);
    ei = min(max(ei, 0), NN - 1);
    tp[(size_t)((ei >> 11) + PAD) * TPW + ((ei & (N_IM - 1)) + PAD)] = (_Float16)1.f;
}

// --------------------------------------------------------- WMMA conv ----
// Block = 256 threads (8 waves). Block tile: 32 rows x 64 cols of output.
// Wave w -> 16x16 tile (tr = w>>2, tc = w&3). im2col GEMM, K = (dy,i),
// 5 chained V_WMMA_F32_16X16X32_F16 per tile (f16 inputs are exact here).
// Patch staging uses the gfx1250 async global->LDS DMA path (ASYNCcnt).
__global__ void __launch_bounds__(256)
rn_conv_kernel(const _Float16* __restrict__ tp,
               const float* __restrict__ kern,
               float* __restrict__ out, unsigned* mm) {
    __shared__ _Float16 spatch[PATCH_H * LDSW];
    __shared__ float    swght[49];
    __shared__ float    sred[512];

    const int X0 = blockIdx.x * 64;
    const int Y0 = blockIdx.y * 32;
    const unsigned tid = threadIdx.x;

    // ---- stage 38x70 halo patch (1330 dwords) global -> LDS, async DMA ----
    {
        unsigned ldsBase = (unsigned)(size_t)&spatch[0];  // AS3 byte offset
        for (int w = (int)tid; w < PATCH_H * PATCH_DW; w += 256) {
            int r  = w / PATCH_DW;
            int cw = w - r * PATCH_DW;
            unsigned ldsOff = ldsBase + (unsigned)(r * (LDSW * 2) + cw * 4);
            unsigned long long ga = (unsigned long long)
                (const void*)(tp + (size_t)(Y0 + r) * TPW + (X0 + 2 * cw));
            asm volatile("global_load_async_to_lds_b32 %0, %1, off"
                         :: "v"(ldsOff), "v"(ga) : "memory");
        }
        if (tid < 49) swght[tid] = kern[tid];  // k[0][0] (all 9 pairs identical)
        asm volatile("s_wait_asynccnt 0x0" ::: "memory");
    }
    __syncthreads();

    const unsigned lane = tid & 31u;
    const int wv  = tid >> 5;
    const int tr  = wv >> 2, tc = wv & 3;
    const int m   = lane & 15;          // A row / C col selector
    const int j   = lane & 15;          // B col
    const int khi = (lane < 16) ? 0 : 8;

    // Build banded weight matrix B per chunk (ISA 16-bit B lane layout)
    V16 Bm[NCHUNK];
#pragma unroll
    for (int cch = 0; cch < NCHUNK; ++cch) {
#pragma unroll
        for (int h = 0; h < 16; ++h) {
            int K  = (h < 8) ? (khi + h) : (16 + khi + h - 8);
            int kk = cch * 32 + K;
            float bv = 0.f;
            if (kk < KTOT) {
                int dy = kk / 22;
                int ii = kk - dy * 22;
                int dx = ii - j;
                if (dx >= 0 && dx < 7) bv = swght[dy * 7 + dx];
            }
            Bm[cch].h[h] = (_Float16)bv;
        }
    }

    v8f acc = {};
#pragma unroll
    for (int cch = 0; cch < NCHUNK; ++cch) {
        V16 Am;
#pragma unroll
        for (int h = 0; h < 16; ++h) {
            int K  = (h < 8) ? (khi + h) : (16 + khi + h - 8);
            int kk = cch * 32 + K;
            int dy = kk / 22;
            int ii = kk - dy * 22;
            int pr = tr * 16 + m + dy;              // kk>=KTOT rows clamped;
            if (pr > PATCH_H - 1) pr = PATCH_H - 1; // B==0 there, so harmless
            Am.h[h] = spatch[pr * LDSW + tc * 16 + ii];
        }
        acc = __builtin_amdgcn_wmma_f32_16x16x32_f16(
            false, Am.v, false, Bm[cch].v, (short)0, acc, false, false);
    }

    // store raw conv (channel 0 region of d_out) + local min/max
    const int orow0 = Y0 + tr * 16 + ((lane < 16) ? 0 : 8);
    const int ocol  = X0 + tc * 16 + j;
    float lmin = 3.402823466e+38f, lmax = 0.f;
#pragma unroll
    for (int v = 0; v < 8; ++v) {
        float val = acc[v];
        out[(size_t)(orow0 + v) * N_IM + ocol] = val;
        lmin = fminf(lmin, val);
        lmax = fmaxf(lmax, val);
    }
    sred[tid]       = lmin;
    sred[256 + tid] = lmax;
    __syncthreads();
    if (tid == 0) {
        float bmin = sred[0], bmax = sred[256];
        for (int t = 1; t < 256; ++t) {
            bmin = fminf(bmin, sred[t]);
            bmax = fmaxf(bmax, sred[256 + t]);
        }
        // conv output >= 0 -> float bits are order-preserving as unsigned
        atomicMin(&mm[0], __float_as_uint(bmin));
        atomicMax(&mm[1], __float_as_uint(bmax));
    }
}

// --------------------------------------------------------- normalize ----
__global__ void rn_norm_kernel(const unsigned* mm, float* out) {
    float mn = __uint_as_float(mm[0]);
    float mx = __uint_as_float(mm[1]);
    float s  = 1.f / (mx - mn);
    long i = (long)blockIdx.x * blockDim.x + threadIdx.x;
    if (i < NN) {
        float v = (out[i] - mn) * s;
        out[i]            = v;
        out[NN + i]       = v;
        out[2L * NN + i]  = v;
    }
}

// ------------------------------------------------------------ launch ----
extern "C" void kernel_launch(void* const* d_in, const int* in_sizes, int n_in,
                              void* d_out, int out_size, void* d_ws, size_t ws_size,
                              hipStream_t stream) {
    const float* x0 = (const float*)d_in[0];
    const float* y0 = (const float*)d_in[1];
    const float* x1 = (const float*)d_in[2];
    const float* y1 = (const float*)d_in[3];
    const float* kr = (const float*)d_in[4];
    float* out = (float*)d_out;

    unsigned*  mm = (unsigned*)d_ws;                      // [min_bits, max_bits]
    _Float16*  tp = (_Float16*)((char*)d_ws + 64);        // padded f16 template

    int initBlocks = (TP_WORDS + 255) / 256;
    rn_init_kernel<<<initBlocks, 256, 0, stream>>>((unsigned*)tp, mm);
    rn_line_kernel<<<1, 1, 0, stream>>>(x0, y0, x1, y1, tp);
    rn_conv_kernel<<<dim3(N_IM / 64, N_IM / 32), 256, 0, stream>>>(tp, kr, out, mm);
    rn_norm_kernel<<<(NN + 255) / 256, 256, 0, stream>>>(mm, out);
}
